// GRCN_17712445129318
// MI455X (gfx1250) — compile-verified, matrix operand order checked by reading, not run on hardware
//
#include <hip/hip_runtime.h>

#define NN 16384
#define FF 256
#define HH 128
#define CC 40
#define KK 32
#define DEGV 16
#define EE (NN*DEGV)
#define NKV (NN*KK)
#define E2V (EE + 2*NKV)
#define CPAD 48

typedef _Float16 v16h  __attribute__((ext_vector_type(16)));
typedef __bf16   v16bf __attribute__((ext_vector_type(16)));
typedef float    v8f   __attribute__((ext_vector_type(8)));

// ---------------- small prep kernels ----------------

__global__ void k_deg0(const float* __restrict__ vals, float* __restrict__ invs0) {
  int i = blockIdx.x * blockDim.x + threadIdx.x;
  if (i < NN) {
    float s = 0.f;
#pragma unroll
    for (int t = 0; t < DEGV; ++t) s += vals[i * DEGV + t];
    invs0[i] = 1.f / (sqrtf(s) + 1e-10f);
  }
}

__global__ void k_norm0(const int* __restrict__ rows, const int* __restrict__ cols,
                        const float* __restrict__ vals, const float* __restrict__ invs,
                        float* __restrict__ nvals) {
  int e = blockIdx.x * blockDim.x + threadIdx.x;
  if (e < EE) nvals[e] = vals[e] * invs[rows[e]] * invs[cols[e]];
}

// SpMM on the sorted, fixed-degree-16 input graph: one block per row, thread j = feature j.
template <bool TANH>
__global__ void k_spmm_diag(const int* __restrict__ cols, const float* __restrict__ nvals,
                            const float* __restrict__ xin, const float* __restrict__ wd,
                            float* __restrict__ outp) {
  int i = blockIdx.x;
  int j = threadIdx.x;  // 256 threads == FF
  float acc = 0.f;
#pragma unroll
  for (int t = 0; t < DEGV; ++t) {
    int e = i * DEGV + t;
    acc += nvals[e] * xin[(size_t)cols[e] * FF + j];
  }
  acc *= wd[j];
  outp[(size_t)i * FF + j] = TANH ? tanhf(acc) : acc;
}

__global__ void k_rownorm_cvt(const float* __restrict__ emb2, _Float16* __restrict__ embh) {
  __shared__ float red[FF];
  int i = blockIdx.x, j = threadIdx.x;
  float v = emb2[(size_t)i * FF + j];
  red[j] = v * v;
  __syncthreads();
  for (int s = FF / 2; s > 0; s >>= 1) {
    if (j < s) red[j] += red[j + s];
    __syncthreads();
  }
  float r = 1.f / fmaxf(sqrtf(red[0]), 1e-12f);
  embh[(size_t)i * FF + j] = (_Float16)(v * r);
}

// ---------------- fused sim = E*E^T + streaming top-K (f16 WMMA) ----------------
// Block = 8 waves; block owns a 16-row tile, waves stride over the 1024 column tiles.
// B fragments for the next column tile are software-pipelined so the 16 global_load_b128
// are in flight while the current tile's 8 WMMAs + top-k insertion execute.
// tile[] and lv/li[] are wave-private: wave-local s_wait_dscnt replaces block barriers.
__global__ void __launch_bounds__(256) k_sim_topk(const _Float16* __restrict__ embh,
                                                  float* __restrict__ topv,
                                                  int* __restrict__ topi) {
  __shared__ float lv[8][16][KK];
  __shared__ int   li[8][16][KK];
  __shared__ float tile[8][16][16];

  const int tid  = threadIdx.x;
  const int w    = tid >> 5;
  const int lane = tid & 31;
  const int rt   = blockIdx.x;          // row tile index
  const int mrow = lane & 15;           // A-row within tile / C-column / B-column
  const int koff = (lane >> 4) * 16;    // lane-group K offset

  for (int s = tid; s < 8 * 16 * KK; s += 256) {
    (&lv[0][0][0])[s] = -3.0e38f;
    (&li[0][0][0])[s] = 0;
  }
  __syncthreads();

  // A fragments: 8 K-steps, per-lane 16 contiguous f16 from one emb row (identical for all waves).
  v16h a[8];
  const _Float16* arow = embh + (size_t)(rt * 16 + mrow) * FF;
#pragma unroll
  for (int ks = 0; ks < 8; ++ks) a[ks] = *(const v16h*)(arow + ks * 32 + koff);

  float curmin  = -3.0e38f;
  int   minslot = 0;
  const int mhi = (lane >> 4) * 8;

  // prologue: B fragments of the first column tile
  v16h bcur[8];
  {
    const _Float16* bb = embh + (size_t)(w * 16 + mrow) * FF + koff;
#pragma unroll
    for (int ks = 0; ks < 8; ++ks) bcur[ks] = *(const v16h*)(bb + ks * 32);
  }

  int ct = w;
  while (true) {
    const bool last = (ct + 8 >= NN / 16);
    v16h bnxt[8];
    if (!last) {
      const _Float16* bb = embh + (size_t)((ct + 8) * 16 + mrow) * FF + koff;
      if (ct + 16 < NN / 16)  // two tiles ahead -> L2/L0 (global_prefetch_b8)
        __builtin_prefetch(embh + (size_t)((ct + 16) * 16 + mrow) * FF + koff, 0, 3);
#pragma unroll
      for (int ks = 0; ks < 8; ++ks) bnxt[ks] = *(const v16h*)(bb + ks * 32);
    }

    v8f acc = {};
#pragma unroll
    for (int ks = 0; ks < 8; ++ks)
      acc = __builtin_amdgcn_wmma_f32_16x16x32_f16(false, a[ks], false, bcur[ks],
                                                   (short)0, acc, false, false);

    // C layout: VGPR vg, lanes 0-15 -> (M=vg, N=lane); lanes 16-31 -> (M=8+vg, N=lane-16)
#pragma unroll
    for (int vg = 0; vg < 8; ++vg) tile[w][mhi + vg][mrow] = acc[vg];
    asm volatile("s_wait_dscnt 0" ::: "memory");  // wave-local: stores -> loads ordering

    if (lane < 16) {
      const int m = lane;
#pragma unroll
      for (int n = 0; n < 16; ++n) {
        float v = tile[w][m][n];
        if (v > curmin) {
          lv[w][m][minslot] = v;
          li[w][m][minslot] = ct * 16 + n;
          float mn = lv[w][m][0];
          int   ms = 0;
#pragma unroll
          for (int s = 1; s < KK; ++s) {
            float q = lv[w][m][s];
            if (q < mn) { mn = q; ms = s; }
          }
          curmin  = mn;
          minslot = ms;
        }
      }
    }
    asm volatile("" ::: "memory");  // keep next tile stores after this iteration's loads

    if (last) break;
#pragma unroll
    for (int ks = 0; ks < 8; ++ks) bcur[ks] = bnxt[ks];
    ct += 8;
  }
  __syncthreads();

  // Merge the 8 per-wave lists into a descending global top-32 (matches top_k order).
  if (tid < 16) {
    int row = rt * 16 + tid;
    for (int k = 0; k < KK; ++k) {
      float best = -3.1e38f;
      int bw = 0, bs = 0;
      for (int ww = 0; ww < 8; ++ww)
        for (int s = 0; s < KK; ++s) {
          float q = lv[ww][tid][s];
          if (q > best) { best = q; bw = ww; bs = s; }
        }
      topv[(size_t)row * KK + k] = best;
      topi[(size_t)row * KK + k] = li[bw][tid][bs];
      lv[bw][tid][bs] = -3.2e38f;
    }
  }
}

// ---------------- graph assembly ----------------

__global__ void k_copy_edges(const int* __restrict__ rows, const int* __restrict__ cols,
                             const float* __restrict__ vals,
                             float* __restrict__ fnr, float* __restrict__ fnc,
                             float* __restrict__ fnv, int* __restrict__ nr, int* __restrict__ nc) {
  int e = blockIdx.x * blockDim.x + threadIdx.x;
  if (e < EE) {
    int r = rows[e], c = cols[e];
    fnr[e] = (float)r; fnc[e] = (float)c; fnv[e] = vals[e];
    nr[e] = r; nc[e] = c;
  }
}

__global__ void k_write_mid(const float* __restrict__ topv, const int* __restrict__ topi,
                            float* __restrict__ fmr, float* __restrict__ fmc, float* __restrict__ fmv,
                            float* __restrict__ fnr, float* __restrict__ fnc, float* __restrict__ fnv,
                            int* __restrict__ nr, int* __restrict__ nc) {
  int t = blockIdx.x * blockDim.x + threadIdx.x;
  if (t < NKV) {
    int i = t / KK;
    int c = topi[t];
    float v = topv[t];
    fmr[t] = (float)i; fmr[NKV + t] = (float)c;
    fmc[t] = (float)c; fmc[NKV + t] = (float)i;
    fmv[t] = v;        fmv[NKV + t] = v;
    fnr[EE + t] = (float)i; fnr[EE + NKV + t] = (float)c;
    fnc[EE + t] = (float)c; fnc[EE + NKV + t] = (float)i;
    fnv[EE + t] = v;        fnv[EE + NKV + t] = v;
    nr[EE + t] = i; nr[EE + NKV + t] = c;
    nc[EE + t] = c; nc[EE + NKV + t] = i;
  }
}

__global__ void k_zero(float* __restrict__ p, long long n) {
  long long stride = (long long)gridDim.x * blockDim.x;
  for (long long i = (long long)blockIdx.x * blockDim.x + threadIdx.x; i < n; i += stride)
    p[i] = 0.f;
}

__global__ void k_deg2(const int* __restrict__ nr, const float* __restrict__ raw,
                       float* __restrict__ deg) {
  int e = blockIdx.x * blockDim.x + threadIdx.x;
  if (e < E2V) atomicAdd(&deg[nr[e]], raw[e]);
}

__global__ void k_invsqrt(float* __restrict__ d) {
  int i = blockIdx.x * blockDim.x + threadIdx.x;
  if (i < NN) d[i] = 1.f / (sqrtf(d[i]) + 1e-10f);
}

__global__ void k_norm2(const int* __restrict__ nr, const int* __restrict__ nc,
                        const float* __restrict__ raw, const float* __restrict__ invs,
                        float* __restrict__ nv) {
  int e = blockIdx.x * blockDim.x + threadIdx.x;
  if (e < E2V) nv[e] = raw[e] * invs[nr[e]] * invs[nc[e]];
}

// Atomic SpMM for the irregular combined graph.
template <int D>
__global__ void k_spmm_atomic(const int* __restrict__ nr, const int* __restrict__ nc,
                              const float* __restrict__ nv, const float* __restrict__ xin,
                              float* __restrict__ outp) {
  long long total  = (long long)E2V * D;
  long long stride = (long long)gridDim.x * blockDim.x;
  for (long long t = (long long)blockIdx.x * blockDim.x + threadIdx.x; t < total; t += stride) {
    int e = (int)(t / D);
    int j = (int)(t % D);
    atomicAdd(&outp[(size_t)nr[e] * D + j], nv[e] * xin[(size_t)nc[e] * D + j]);
  }
}

__global__ void k_f32_to_bf16(const float* __restrict__ a, __bf16* __restrict__ b, long long n) {
  long long stride = (long long)gridDim.x * blockDim.x;
  for (long long i = (long long)blockIdx.x * blockDim.x + threadIdx.x; i < n; i += stride)
    b[i] = (__bf16)a[i];
}

__global__ void k_tr_w1(const float* __restrict__ Wg1, __bf16* __restrict__ wt) {
  int t = blockIdx.x * blockDim.x + threadIdx.x;  // FF*HH
  if (t < FF * HH) {
    int k = t / HH, n = t % HH;
    wt[(size_t)n * FF + k] = (__bf16)Wg1[t];
  }
}

__global__ void k_tr_w2(const float* __restrict__ Wg2, __bf16* __restrict__ wt) {
  int t = blockIdx.x * blockDim.x + threadIdx.x;  // CPAD*HH, wt layout [CPAD][HH]
  if (t < CPAD * HH) {
    int n = t / HH, k = t % HH;
    wt[t] = (n < CC) ? (__bf16)Wg2[(size_t)k * CC + n] : (__bf16)0.f;
  }
}

// ---------------- task-GCN GEMMs (bf16 WMMA) ----------------

// h1 = relu(t1 @ Wg1 + bg1). Grid: N/16 row tiles; 8 waves = 8 column tiles (HH/16).
__global__ void __launch_bounds__(256) k_gemm1(const __bf16* __restrict__ A,
                                               const __bf16* __restrict__ BT,
                                               const float* __restrict__ bias,
                                               float* __restrict__ outp) {
  int tid = threadIdx.x, w = tid >> 5, lane = tid & 31;
  int bt = blockIdx.x, ct = w;
  int mrow = lane & 15, koff = (lane >> 4) * 16;
  const __bf16* arow = A  + (size_t)(bt * 16 + mrow) * FF + koff;
  const __bf16* brow = BT + (size_t)(ct * 16 + mrow) * FF + koff;
  v8f acc = {};
#pragma unroll
  for (int ks = 0; ks < FF / 32; ++ks) {
    v16bf a = *(const v16bf*)(arow + ks * 32);
    v16bf b = *(const v16bf*)(brow + ks * 32);
    acc = __builtin_amdgcn_wmma_f32_16x16x32_bf16(false, a, false, b, (short)0, acc, false, false);
  }
  int mhi = (lane >> 4) * 8;
#pragma unroll
  for (int vg = 0; vg < 8; ++vg) {
    int m = bt * 16 + mhi + vg;
    int n = ct * 16 + mrow;
    outp[(size_t)m * HH + n] = fmaxf(acc[vg] + bias[n], 0.f);
  }
}

// out = t2 @ Wg2 + bg2 (C=40 padded to 48 -> 3 waves/block).
__global__ void __launch_bounds__(96) k_gemm2(const __bf16* __restrict__ A,
                                              const __bf16* __restrict__ BT,
                                              const float* __restrict__ bias,
                                              float* __restrict__ outp) {
  int tid = threadIdx.x, w = tid >> 5, lane = tid & 31;
  int bt = blockIdx.x, ct = w;  // 0..2
  int mrow = lane & 15, koff = (lane >> 4) * 16;
  const __bf16* arow = A  + (size_t)(bt * 16 + mrow) * HH + koff;
  const __bf16* brow = BT + (size_t)(ct * 16 + mrow) * HH + koff;
  v8f acc = {};
#pragma unroll
  for (int ks = 0; ks < HH / 32; ++ks) {
    v16bf a = *(const v16bf*)(arow + ks * 32);
    v16bf b = *(const v16bf*)(brow + ks * 32);
    acc = __builtin_amdgcn_wmma_f32_16x16x32_bf16(false, a, false, b, (short)0, acc, false, false);
  }
  int mhi = (lane >> 4) * 8;
#pragma unroll
  for (int vg = 0; vg < 8; ++vg) {
    int n = ct * 16 + mrow;
    if (n < CC)
      outp[(size_t)(bt * 16 + mhi + vg) * CC + n] = acc[vg] + bias[n];
  }
}

// ---------------- driver ----------------

extern "C" void kernel_launch(void* const* d_in, const int* in_sizes, int n_in,
                              void* d_out, int out_size, void* d_ws, size_t ws_size,
                              hipStream_t stream) {
  (void)in_sizes; (void)n_in; (void)out_size; (void)ws_size;
  const float* x     = (const float*)d_in[0];
  const int*   adji  = (const int*)d_in[1];
  const float* avals = (const float*)d_in[2];
  const float* Wd1   = (const float*)d_in[3];
  const float* Wd2   = (const float*)d_in[4];
  const float* Wg1   = (const float*)d_in[5];
  const float* bg1   = (const float*)d_in[6];
  const float* Wg2   = (const float*)d_in[7];
  const float* bg2   = (const float*)d_in[8];
  const int* rows = adji;
  const int* cols = adji + EE;

  float* f_out  = (float*)d_out;                       // [N,C]
  float* f_midr = f_out + (size_t)NN * CC;             // mid_indices row 0 (2NK)
  float* f_midc = f_midr + 2 * (size_t)NKV;            // mid_indices row 1
  float* f_midv = f_midc + 2 * (size_t)NKV;            // mid_vals (2NK)
  float* f_newr = f_midv + 2 * (size_t)NKV;            // new_indices row 0 (E2)
  float* f_newc = f_newr + (size_t)E2V;                // new_indices row 1
  float* f_newv = f_newc + (size_t)E2V;                // new_vals (E2)

  char* wp = (char*)d_ws;
  auto carve = [&wp](size_t bytes) -> void* {
    void* p = (void*)wp;
    wp += (bytes + 255) & ~(size_t)255;
    return p;
  };
  float*    invs0  = (float*)carve((size_t)NN * 4);
  float*    nvals0 = (float*)carve((size_t)EE * 4);
  float*    emb    = (float*)carve((size_t)NN * FF * 4);
  float*    emb2   = (float*)carve((size_t)NN * FF * 4);
  _Float16* embh   = (_Float16*)carve((size_t)NN * FF * 2);
  float*    topv   = (float*)carve((size_t)NKV * 4);
  int*      topi   = (int*)carve((size_t)NKV * 4);
  int*      nr     = (int*)carve((size_t)E2V * 4);
  int*      nc     = (int*)carve((size_t)E2V * 4);
  float*    nvals2 = (float*)carve((size_t)E2V * 4);
  float*    invs1  = (float*)carve((size_t)NN * 4);
  __bf16*   t1b    = (__bf16*)carve((size_t)NN * FF * 2);
  __bf16*   wg1t   = (__bf16*)carve((size_t)HH * FF * 2);
  __bf16*   t2b    = (__bf16*)carve((size_t)NN * HH * 2);
  __bf16*   wg2t   = (__bf16*)carve((size_t)CPAD * HH * 2);
  float*    t1 = emb2;                       // reuse: emb2 dead after row-norm
  float*    h1 = emb;                        // reuse: emb dead after 2nd diag conv
  float*    t2 = emb + (size_t)NN * HH;

  // graph branch
  k_deg0<<<NN / 256, 256, 0, stream>>>(avals, invs0);
  k_norm0<<<EE / 256, 256, 0, stream>>>(rows, cols, avals, invs0, nvals0);
  k_spmm_diag<true><<<NN, FF, 0, stream>>>(cols, nvals0, x, Wd1, emb);
  k_spmm_diag<false><<<NN, FF, 0, stream>>>(cols, nvals0, emb, Wd2, emb2);
  k_rownorm_cvt<<<NN, FF, 0, stream>>>(emb2, embh);

  // fused similarity GEMM + top-K (f16 WMMA, emb resident in L2)
  k_sim_topk<<<NN / 16, 256, 0, stream>>>(embh, topv, topi);

  // assemble symmetrized residual graph
  k_copy_edges<<<EE / 256, 256, 0, stream>>>(rows, cols, avals, f_newr, f_newc, f_newv, nr, nc);
  k_write_mid<<<NKV / 256, 256, 0, stream>>>(topv, topi, f_midr, f_midc, f_midv,
                                             f_newr, f_newc, f_newv, nr, nc);
  k_zero<<<64, 256, 0, stream>>>(invs1, NN);
  k_deg2<<<E2V / 256, 256, 0, stream>>>(nr, f_newv, invs1);
  k_invsqrt<<<NN / 256, 256, 0, stream>>>(invs1);
  k_norm2<<<E2V / 256, 256, 0, stream>>>(nr, nc, f_newv, invs1, nvals2);

  // task GCN layer 1
  k_zero<<<4096, 256, 0, stream>>>(t1, (long long)NN * FF);
  k_spmm_atomic<FF><<<32768, 256, 0, stream>>>(nr, nc, nvals2, x, t1);
  k_f32_to_bf16<<<8192, 256, 0, stream>>>(t1, t1b, (long long)NN * FF);
  k_tr_w1<<<(FF * HH) / 256, 256, 0, stream>>>(Wg1, wg1t);
  k_gemm1<<<NN / 16, 256, 0, stream>>>(t1b, wg1t, bg1, h1);

  // task GCN layer 2
  k_zero<<<2048, 256, 0, stream>>>(t2, (long long)NN * HH);
  k_spmm_atomic<HH><<<32768, 256, 0, stream>>>(nr, nc, nvals2, h1, t2);
  k_f32_to_bf16<<<4096, 256, 0, stream>>>(t2, t2b, (long long)NN * HH);
  k_tr_w2<<<(CPAD * HH) / 256, 256, 0, stream>>>(Wg2, wg2t);
  k_gemm2<<<NN / 16, 96, 0, stream>>>(t2b, wg2t, bg2, f_out);
}
